// MegNetLayer_83708912599380
// MI455X (gfx1250) — matrix operand order repaired and checked by reading, not compiled, not run on hardware
//
#include <hip/hip_runtime.h>
#include <hip/hip_bf16.h>

typedef _Float16 f16_t;
typedef __attribute__((ext_vector_type(16))) _Float16 v16h;
typedef __attribute__((ext_vector_type(8)))  _Float16 v8h;
typedef __attribute__((ext_vector_type(8)))  float    v8f;

#define WMMA_F16(a, b, c) \
  __builtin_amdgcn_wmma_f32_16x16x32_f16(false, (a), false, (b), (short)0, (c), false, false)

// Branchless SELU: scale*max(x,0) + scale*alpha*(exp(min(x,0)) - 1).
// exp(min(x,0))-1 == 0 for x>=0, so this is exact and emits no branches.
__device__ __forceinline__ float selu_f(float x) {
  const float kScale      = 1.0507009873554805f;
  const float kScaleAlpha = 1.7580993408473766f;  // scale * alpha
  return kScale * fmaxf(x, 0.f) + kScaleAlpha * (__expf(fminf(x, 0.f)) - 1.f);
}

__device__ __forceinline__ v8f splat8(float x) {
  v8f v;
#pragma unroll
  for (int i = 0; i < 8; ++i) v[i] = x;
  return v;
}

// ---------------------------------------------------------------------------
// Weight staging: convert f32 weights [din x dout] (row-major, x@W) into LDS
// in *fragment order*: frag f = kt*ntiles+nt, 32 lane-slots x 16 halves
// contiguous.  B layout (contiguous k, as load_afrag): half h = K(kt*32 + 16*(lane>=16) + h),
// column N = nt*16 + lane%16.  Done once per block, shared by all 8 waves.
// ---------------------------------------------------------------------------
__device__ void stage_wfrags(f16_t* dst, const float* w, int dout, int ktiles, int ntiles) {
  const int nSlots = ktiles * ntiles * 32;
  for (int idx = threadIdx.x; idx < nSlots; idx += blockDim.x) {
    const int f  = idx >> 5;
    const int l  = idx & 31;
    const int kt = f / ntiles;
    const int nt = f - kt * ntiles;
    const int n  = nt * 16 + (l & 15);
    const int k0 = kt * 32 + (l >> 4) * 16;
    f16_t* o = dst + idx * 16;
#pragma unroll
    for (int i = 0; i < 16; ++i) o[i] = (f16_t)w[(k0 + i) * dout + n];
  }
}

// B-fragment: contiguous 32B per lane -> two ds_load_b128.
__device__ __forceinline__ v16h load_bfrag(const f16_t* Wf, int frag, int lane) {
  const v8h* p = (const v8h*)(Wf + (frag * 32 + lane) * 16);
  v8h lo = p[0];
  v8h up = p[1];
  return __builtin_shufflevector(lo, up, 0, 1, 2, 3, 4, 5, 6, 7, 8, 9, 10, 11, 12, 13, 14, 15);
}

// A-fragment 16x32 f16 from LDS row-major activations (stride in halves).
__device__ __forceinline__ v16h load_afrag(const f16_t* X, int stride, int kt, int lane) {
  const int m  = lane & 15;
  const int hi = lane >> 4;
  const v8h* p = (const v8h*)(X + m * stride + kt * 32 + hi * 16);
  v8h lo = p[0];
  v8h up = p[1];  // +8 halves
  return __builtin_shufflevector(lo, up, 0, 1, 2, 3, 4, 5, 6, 7, 8, 9, 10, 11, 12, 13, 14, 15);
}

// SELU + f32->f16, C-fragment -> LDS row-major (row m = r + 8*(lane>=16), col n).
__device__ __forceinline__ void store_hfrag(v8f c, f16_t* H, int stride, int nt, int lane) {
  const int lane15 = lane & 15;
  const int hi     = lane >> 4;
  const int n = nt * 16 + lane15;
#pragma unroll
  for (int r = 0; r < 8; ++r) H[(r + 8 * hi) * stride + n] = (f16_t)selu_f(c[r]);
}

#define XS_E 136  // 128 + pad (halves)
#define XS_V 104  // 96 + pad
#define HS_M 72   // 64 + pad

// ---------------------------------------------------------------------------
// Kernel 1: phi_e over bonds. Fuses: gather atom feats, 128->64->64->32 SELU
// MLP (WMMA f16, f32 accum), write bonds_new, scatter-add b2a/cnt, bond-mean
// partials. One wave per 16-bond tile, grid-stride.
// ---------------------------------------------------------------------------
__global__ __launch_bounds__(256) void phi_e_kernel(
    const float* __restrict__ bonds, const int* __restrict__ ba1, const int* __restrict__ ba2,
    const float* __restrict__ atoms, const float* __restrict__ state,
    const float* __restrict__ w1, const float* __restrict__ b1,
    const float* __restrict__ w2, const float* __restrict__ b2,
    const float* __restrict__ w3, const float* __restrict__ b3,
    float* __restrict__ bonds_new, float* __restrict__ b2a_sum,
    float* __restrict__ cnt, float* __restrict__ bsum, int nBonds) {
  __shared__ alignas(16) f16_t sW1[16 * 32 * 16];  // 4kt x 4nt frags
  __shared__ alignas(16) f16_t sW2[8 * 32 * 16];   // 2kt x 4nt
  __shared__ alignas(16) f16_t sW3[4 * 32 * 16];   // 2kt x 2nt
  __shared__ alignas(16) f16_t sX[8 * 16 * XS_E];
  __shared__ alignas(16) f16_t sH[8 * 16 * HS_M];

  stage_wfrags(sW1, w1, 64, 4, 4);
  stage_wfrags(sW2, w2, 64, 2, 4);
  stage_wfrags(sW3, w3, 32, 2, 2);
  __syncthreads();

  const int lane   = threadIdx.x & 31;
  const int wid    = threadIdx.x >> 5;
  const int lane15 = lane & 15;
  const int hi     = lane >> 4;
  f16_t* X = sX + wid * 16 * XS_E;
  f16_t* H = sH + wid * 16 * HS_M;

  float bias1[4], bias2[4], bias3[2];
#pragma unroll
  for (int nt = 0; nt < 4; ++nt) { bias1[nt] = b1[nt * 16 + lane15]; bias2[nt] = b2[nt * 16 + lane15]; }
#pragma unroll
  for (int nt = 0; nt < 2; ++nt) bias3[nt] = b3[nt * 16 + lane15];
  const f16_t stateH = (f16_t)state[lane];

  float colsum0 = 0.f, colsum1 = 0.f;
  const int tiles  = nBonds >> 4;  // N_BONDS divisible by 16
  const int nWaves = (gridDim.x * blockDim.x) >> 5;
  const int gw     = (blockIdx.x * blockDim.x + threadIdx.x) >> 5;

  for (int t = gw; t < tiles; t += nWaves) {
    const int e0 = t << 4;
    int i1save[8];  // atom-1 index for the 8 rows this lane owns in the epilogue
    // Stage x = [a1 | a2 | bond | state] as f16 into LDS (lane = feature col).
#pragma unroll 4
    for (int m = 0; m < 16; ++m) {
      const int e  = e0 + m;
      const int i1 = ba1[e];
      const int i2 = ba2[e];
      if ((m >> 3) == hi) i1save[m & 7] = i1;
      X[m * XS_E +      lane] = (f16_t)atoms[(size_t)i1 * 32 + lane];
      X[m * XS_E + 32 + lane] = (f16_t)atoms[(size_t)i2 * 32 + lane];
      X[m * XS_E + 64 + lane] = (f16_t)bonds[(size_t)e  * 32 + lane];
      X[m * XS_E + 96 + lane] = stateH;
    }
    // (same-wave LDS ops are in-order on CDNA5: no barrier needed)

    // Layer 1: 128 -> 64
    v8f c1[4];
#pragma unroll
    for (int nt = 0; nt < 4; ++nt) c1[nt] = splat8(bias1[nt]);
#pragma unroll
    for (int kt = 0; kt < 4; ++kt) {
      v16h a = load_afrag(X, XS_E, kt, lane);
#pragma unroll
      for (int nt = 0; nt < 4; ++nt)
        c1[nt] = WMMA_F16(a, load_bfrag(sW1, kt * 4 + nt, lane), c1[nt]);
    }
#pragma unroll
    for (int nt = 0; nt < 4; ++nt) store_hfrag(c1[nt], H, HS_M, nt, lane);

    // Layer 2: 64 -> 64
    v8f c2[4];
#pragma unroll
    for (int nt = 0; nt < 4; ++nt) c2[nt] = splat8(bias2[nt]);
#pragma unroll
    for (int kt = 0; kt < 2; ++kt) {
      v16h a = load_afrag(H, HS_M, kt, lane);
#pragma unroll
      for (int nt = 0; nt < 4; ++nt)
        c2[nt] = WMMA_F16(a, load_bfrag(sW2, kt * 4 + nt, lane), c2[nt]);
    }
#pragma unroll
    for (int nt = 0; nt < 4; ++nt) store_hfrag(c2[nt], H, HS_M, nt, lane);

    // Layer 3: 64 -> 32
    v8f c3[2];
#pragma unroll
    for (int nt = 0; nt < 2; ++nt) c3[nt] = splat8(bias3[nt]);
#pragma unroll
    for (int kt = 0; kt < 2; ++kt) {
      v16h a = load_afrag(H, HS_M, kt, lane);
#pragma unroll
      for (int nt = 0; nt < 2; ++nt)
        c3[nt] = WMMA_F16(a, load_bfrag(sW3, kt * 2 + nt, lane), c3[nt]);
    }

    // SELU, write bonds_new, scatter-add to atom1, accumulate mean partials.
#pragma unroll
    for (int r = 0; r < 8; ++r) {
      const int e  = e0 + r + 8 * hi;
      const int i1 = i1save[r];
      const float v0 = selu_f(c3[0][r]);
      const float v1 = selu_f(c3[1][r]);
      bonds_new[(size_t)e * 32 +      lane15] = v0;
      bonds_new[(size_t)e * 32 + 16 + lane15] = v1;
      atomicAdd(&b2a_sum[(size_t)i1 * 32 +      lane15], v0);
      atomicAdd(&b2a_sum[(size_t)i1 * 32 + 16 + lane15], v1);
      colsum0 += v0;
      colsum1 += v1;
    }
    if (lane < 16) atomicAdd(&cnt[ba1[e0 + lane]], 1.0f);
  }
  atomicAdd(&bsum[lane15],      colsum0);
  atomicAdd(&bsum[16 + lane15], colsum1);
}

// ---------------------------------------------------------------------------
// Kernel 2: phi_v over atoms: x = [b2a/cnt | atoms | state] -> 96->64->64->32.
// ---------------------------------------------------------------------------
__global__ __launch_bounds__(256) void phi_v_kernel(
    const float* __restrict__ b2a_sum, const float* __restrict__ cnt,
    const float* __restrict__ atoms, const float* __restrict__ state,
    const float* __restrict__ w1, const float* __restrict__ b1,
    const float* __restrict__ w2, const float* __restrict__ b2,
    const float* __restrict__ w3, const float* __restrict__ b3,
    float* __restrict__ atoms_new, float* __restrict__ asum, int nAtoms) {
  __shared__ alignas(16) f16_t sW1[12 * 32 * 16];  // 3kt x 4nt
  __shared__ alignas(16) f16_t sW2[8 * 32 * 16];   // 2kt x 4nt
  __shared__ alignas(16) f16_t sW3[4 * 32 * 16];   // 2kt x 2nt
  __shared__ alignas(16) f16_t sX[8 * 16 * XS_V];
  __shared__ alignas(16) f16_t sH[8 * 16 * HS_M];

  stage_wfrags(sW1, w1, 64, 3, 4);
  stage_wfrags(sW2, w2, 64, 2, 4);
  stage_wfrags(sW3, w3, 32, 2, 2);
  __syncthreads();

  const int lane   = threadIdx.x & 31;
  const int wid    = threadIdx.x >> 5;
  const int lane15 = lane & 15;
  const int hi     = lane >> 4;
  f16_t* X = sX + wid * 16 * XS_V;
  f16_t* H = sH + wid * 16 * HS_M;

  float bias1[4], bias2[4], bias3[2];
#pragma unroll
  for (int nt = 0; nt < 4; ++nt) { bias1[nt] = b1[nt * 16 + lane15]; bias2[nt] = b2[nt * 16 + lane15]; }
#pragma unroll
  for (int nt = 0; nt < 2; ++nt) bias3[nt] = b3[nt * 16 + lane15];
  const f16_t stateH = (f16_t)state[lane];

  float colsum0 = 0.f, colsum1 = 0.f;
  const int tiles  = nAtoms >> 4;  // N_ATOMS divisible by 16
  const int nWaves = (gridDim.x * blockDim.x) >> 5;
  const int gw     = (blockIdx.x * blockDim.x + threadIdx.x) >> 5;

  for (int t = gw; t < tiles; t += nWaves) {
    const int i0 = t << 4;
#pragma unroll 4
    for (int m = 0; m < 16; ++m) {
      const int i = i0 + m;
      const float c = cnt[i];
      X[m * XS_V +      lane] = (f16_t)(b2a_sum[(size_t)i * 32 + lane] / c);
      X[m * XS_V + 32 + lane] = (f16_t)atoms[(size_t)i * 32 + lane];
      X[m * XS_V + 64 + lane] = stateH;
    }

    v8f c1[4];
#pragma unroll
    for (int nt = 0; nt < 4; ++nt) c1[nt] = splat8(bias1[nt]);
#pragma unroll
    for (int kt = 0; kt < 3; ++kt) {
      v16h a = load_afrag(X, XS_V, kt, lane);
#pragma unroll
      for (int nt = 0; nt < 4; ++nt)
        c1[nt] = WMMA_F16(a, load_bfrag(sW1, kt * 4 + nt, lane), c1[nt]);
    }
#pragma unroll
    for (int nt = 0; nt < 4; ++nt) store_hfrag(c1[nt], H, HS_M, nt, lane);

    v8f c2[4];
#pragma unroll
    for (int nt = 0; nt < 4; ++nt) c2[nt] = splat8(bias2[nt]);
#pragma unroll
    for (int kt = 0; kt < 2; ++kt) {
      v16h a = load_afrag(H, HS_M, kt, lane);
#pragma unroll
      for (int nt = 0; nt < 4; ++nt)
        c2[nt] = WMMA_F16(a, load_bfrag(sW2, kt * 4 + nt, lane), c2[nt]);
    }
#pragma unroll
    for (int nt = 0; nt < 4; ++nt) store_hfrag(c2[nt], H, HS_M, nt, lane);

    v8f c3[2];
#pragma unroll
    for (int nt = 0; nt < 2; ++nt) c3[nt] = splat8(bias3[nt]);
#pragma unroll
    for (int kt = 0; kt < 2; ++kt) {
      v16h a = load_afrag(H, HS_M, kt, lane);
#pragma unroll
      for (int nt = 0; nt < 2; ++nt)
        c3[nt] = WMMA_F16(a, load_bfrag(sW3, kt * 2 + nt, lane), c3[nt]);
    }

#pragma unroll
    for (int r = 0; r < 8; ++r) {
      const int i = i0 + r + 8 * hi;
      const float v0 = selu_f(c3[0][r]);
      const float v1 = selu_f(c3[1][r]);
      atoms_new[(size_t)i * 32 +      lane15] = v0;
      atoms_new[(size_t)i * 32 + 16 + lane15] = v1;
      colsum0 += v0;
      colsum1 += v1;
    }
  }
  atomicAdd(&asum[lane15],      colsum0);
  atomicAdd(&asum[16 + lane15], colsum1);
}

// ---------------------------------------------------------------------------
// Kernel 3: phi_u on one row: x = [mean(bonds_new)|mean(atoms_new)|state]
// -> 96->64->64->32. Negligible cost: plain VALU, one block.
// ---------------------------------------------------------------------------
__global__ __launch_bounds__(64) void phi_u_kernel(
    const float* __restrict__ state, const float* __restrict__ bsum, const float* __restrict__ asum,
    const float* __restrict__ w1, const float* __restrict__ b1,
    const float* __restrict__ w2, const float* __restrict__ b2,
    const float* __restrict__ w3, const float* __restrict__ b3,
    float* __restrict__ state_new, float invE, float invN) {
  __shared__ float x[96], h1[64], h2[64];
  const int t = threadIdx.x;
  if (t < 32) {
    x[t]      = bsum[t] * invE;
    x[32 + t] = asum[t] * invN;
    x[64 + t] = state[t];
  }
  __syncthreads();
  float acc = b1[t];
  for (int k = 0; k < 96; ++k) acc += x[k] * w1[k * 64 + t];
  h1[t] = selu_f(acc);
  __syncthreads();
  acc = b2[t];
  for (int k = 0; k < 64; ++k) acc += h1[k] * w2[k * 64 + t];
  h2[t] = selu_f(acc);
  __syncthreads();
  if (t < 32) {
    acc = b3[t];
    for (int k = 0; k < 64; ++k) acc += h2[k] * w3[k * 32 + t];
    state_new[t] = selu_f(acc);
  }
}

extern "C" void kernel_launch(void* const* d_in, const int* in_sizes, int n_in,
                              void* d_out, int out_size, void* d_ws, size_t ws_size,
                              hipStream_t stream) {
  (void)n_in; (void)out_size; (void)ws_size;
  const float* bonds = (const float*)d_in[0];
  const int*   ba1   = (const int*)d_in[1];
  const int*   ba2   = (const int*)d_in[2];
  const float* atoms = (const float*)d_in[3];
  const float* state = (const float*)d_in[4];
  const float* e_w1 = (const float*)d_in[5];  const float* e_b1 = (const float*)d_in[6];
  const float* e_w2 = (const float*)d_in[7];  const float* e_b2 = (const float*)d_in[8];
  const float* e_w3 = (const float*)d_in[9];  const float* e_b3 = (const float*)d_in[10];
  const float* v_w1 = (const float*)d_in[11]; const float* v_b1 = (const float*)d_in[12];
  const float* v_w2 = (const float*)d_in[13]; const float* v_b2 = (const float*)d_in[14];
  const float* v_w3 = (const float*)d_in[15]; const float* v_b3 = (const float*)d_in[16];
  const float* u_w1 = (const float*)d_in[17]; const float* u_b1 = (const float*)d_in[18];
  const float* u_w2 = (const float*)d_in[19]; const float* u_b2 = (const float*)d_in[20];
  const float* u_w3 = (const float*)d_in[21]; const float* u_b3 = (const float*)d_in[22];

  const int nBonds = in_sizes[0] / 32;
  const int nAtoms = in_sizes[3] / 32;

  float* out       = (float*)d_out;
  float* bonds_new = out;
  float* atoms_new = out + (size_t)nBonds * 32;
  float* state_new = atoms_new + (size_t)nAtoms * 32;

  // Workspace layout (f32): b2a_sum[nAtoms*32] | cnt[nAtoms] | bsum[32] | asum[32]
  float* ws      = (float*)d_ws;
  float* b2a_sum = ws;
  float* cnt     = ws + (size_t)nAtoms * 32;
  float* bsum    = cnt + nAtoms;
  float* asum    = bsum + 32;
  const size_t zero_bytes = ((size_t)nAtoms * 33 + 64) * sizeof(float);
  hipMemsetAsync(d_ws, 0, zero_bytes, stream);

  phi_e_kernel<<<1024, 256, 0, stream>>>(bonds, ba1, ba2, atoms, state,
                                         e_w1, e_b1, e_w2, e_b2, e_w3, e_b3,
                                         bonds_new, b2a_sum, cnt, bsum, nBonds);
  phi_v_kernel<<<256, 256, 0, stream>>>(b2a_sum, cnt, atoms, state,
                                        v_w1, v_b1, v_w2, v_b2, v_w3, v_b3,
                                        atoms_new, asum, nAtoms);
  phi_u_kernel<<<1, 64, 0, stream>>>(state, bsum, asum,
                                     u_w1, u_b1, u_w2, u_b2, u_w3, u_b3,
                                     state_new, 1.0f / (float)nBonds, 1.0f / (float)nAtoms);
}
